// PAdicTemporalBlock_82411832475874
// MI455X (gfx1250) — matrix operand order repaired and checked by reading, not compiled, not run on hardware
//
#include <hip/hip_runtime.h>
#include <hip/hip_bf16.h>
#include <math.h>

// Problem constants (from reference)
#define B_    32
#define T_    243
#define J_    17
#define D_    128       // tangent dim
#define DD_   129       // ambient dim
#define M_    15        // FNO modes
#define HID_  512       // FFN hidden
#define BJ_   (B_*J_)           // 544
#define ROWS_ (B_*T_*J_)        // 132192  (divisible by 16)

typedef unsigned int       u32;
typedef unsigned long long u64;
typedef __attribute__((ext_vector_type(16))) __bf16 v16bf;
typedef __attribute__((ext_vector_type(8)))  float  v8f;
typedef __attribute__((ext_vector_type(4)))  u32    v4u;
typedef __attribute__((ext_vector_type(8)))  int    v8i;
typedef __attribute__((ext_vector_type(4)))  int    v4i;

// ---- WMMA fragment helpers (CDNA5 16x16x32 bf16 layouts, ISA 7.12.2) ----
// A (16x32): lane m = lane&15, half = lane>>4.
//   a[j]   = A[m][k0 + half*8 + j]        (j = 0..7)
//   a[8+j] = A[m][k0 + 16 + half*8 + j]
__device__ __forceinline__ v16bf a_frag(const __bf16* base, int ld, int m, int half, int k0) {
  v16bf a;
  const __bf16* r = base + m * ld + k0 + half * 8;
#pragma unroll
  for (int j = 0; j < 8; ++j) { a[j] = r[j]; a[8 + j] = r[16 + j]; }
  return a;
}
// B (32x16): lane column n = n0 + (lane&15); b[j] = B[k0+half*8+j][n], b[8+j] = B[k0+16+half*8+j][n].
__device__ __forceinline__ v16bf b_frag(const __bf16* Bm, int ldb, int k, int n0) {
  v16bf b;
  int l = k & 31;
  const __bf16* r = Bm + (size_t)((k & ~31) + (l >> 4) * 8) * ldb + n0 + (l & 15);
#pragma unroll
  for (int j = 0; j < 8; ++j) { b[j] = r[(size_t)j * ldb]; b[8 + j] = r[(size_t)(16 + j) * ldb]; }
  return b;
}

// ---- Tensor Data Mover: 2D bf16 tile (global, row-major) -> LDS ----------
// D# per CDNA5 ISA:
//   group0: [1:0]=count=1 | [63:32]=lds_addr | [120:64]=global_addr | [127:126]=type(2)
//   group1: [17:16]=data_size(1 => 2B) | [79:48]=tensor_dim0 | [111:80]=tensor_dim1
//           [127:112]=tile_dim0 | [143:128]=tile_dim1 | [207:160]=tensor_dim0_stride
// 2D tile => groups 2/3 zero.
__device__ __forceinline__ void tdm_load_2d_bf16(const void* gaddr, u32 lds_byte_off,
                                                 u32 tensor_d0, u32 tensor_d1,
                                                 u32 stride_d0, u32 tile_d0, u32 tile_d1) {
  u64 ga = (u64)(uintptr_t)gaddr;
  v4u g0;
  g0[0] = 1u;                                               // count=1 (valid user descriptor)
  g0[1] = lds_byte_off;                                     // lds_addr (bytes)
  g0[2] = (u32)(ga & 0xFFFFFFFFu);                          // global_addr[31:0]
  g0[3] = (u32)((ga >> 32) & 0x01FFFFFFu) | (2u << 30);     // global_addr[56:32] | type=2
  v8i g1;
  g1[0] = (int)(1u << 16);                                  // data_size = 1 (2 bytes/elem)
  g1[1] = (int)((tensor_d0 & 0xFFFFu) << 16);               // tensor_dim0[15:0]
  g1[2] = (int)(((tensor_d0 >> 16) & 0xFFFFu) | ((tensor_d1 & 0xFFFFu) << 16));
  g1[3] = (int)(((tensor_d1 >> 16) & 0xFFFFu) | ((tile_d0 & 0xFFFFu) << 16));
  g1[4] = (int)(tile_d1 & 0xFFFFu);                         // tile_dim1 (tile_dim2 = 0)
  g1[5] = (int)stride_d0;                                   // tensor_dim0_stride[31:0]
  g1[6] = 0;                                                // stride[47:32], dim1_stride lo
  g1[7] = 0;
  v4i z = {0, 0, 0, 0};
#if __clang_major__ >= 23
  v8i z8 = {0, 0, 0, 0, 0, 0, 0, 0};
  __builtin_amdgcn_tensor_load_to_lds(g0, g1, z, z, z8, 0);
#else
  __builtin_amdgcn_tensor_load_to_lds(g0, g1, z, z, 0);
#endif
}

__device__ __forceinline__ u32 lds_offset_of(const void* p) {
  return (u32)(uintptr_t)p;   // low 32 bits of flat LDS aperture address = LDS byte offset
}

// =====================================================================
// K0: weight conversion / relayout
// =====================================================================
__global__ void convert_weights_kernel(const double* __restrict__ wr, const double* __restrict__ wi,
                                       const double* __restrict__ bw, const double* __restrict__ bb,
                                       const float* __restrict__ fc1, const float* __restrict__ fc2,
                                       float* __restrict__ Wr, float* __restrict__ Wi,
                                       __bf16* __restrict__ bwT, float* __restrict__ bbf,
                                       __bf16* __restrict__ w1, __bf16* __restrict__ w2) {
  const int S1 = M_ * D_ * D_;     // Wr
  const int S2 = S1;               // Wi
  const int S3 = D_ * D_;          // bwT
  const int S4 = D_;               // bb
  const int S5 = D_ * HID_;        // fc1
  const int S6 = HID_ * D_;        // fc2
  const int N  = S1 + S2 + S3 + S4 + S5 + S6;
  for (int g = blockIdx.x * blockDim.x + threadIdx.x; g < N; g += gridDim.x * blockDim.x) {
    int idx = g;
    if (idx < S1) {
      int m = idx / (D_ * D_); int r = idx % (D_ * D_); int i = r / D_, o = r % D_;
      Wr[idx] = (float)wr[((size_t)i * D_ + o) * M_ + m];
      continue;
    }
    idx -= S1;
    if (idx < S2) {
      int m = idx / (D_ * D_); int r = idx % (D_ * D_); int i = r / D_, o = r % D_;
      Wi[idx] = (float)wi[((size_t)i * D_ + o) * M_ + m];
      continue;
    }
    idx -= S2;
    if (idx < S3) {
      int k = idx / D_, n = idx % D_;
      bwT[idx] = (__bf16)(float)bw[(size_t)n * D_ + k];   // B[k][n] = bw[n][k]
      continue;
    }
    idx -= S3;
    if (idx < S4) { bbf[idx] = (float)bb[idx]; continue; }
    idx -= S4;
    if (idx < S5) { w1[idx] = (__bf16)fc1[idx]; continue; }
    idx -= S5;
    w2[idx] = (__bf16)fc2[idx];
  }
}

// =====================================================================
// K1: log_map0 -> x_tan[row][0..127], one wave32 per row
// =====================================================================
__global__ void logmap_kernel(const float* __restrict__ xseq, float* __restrict__ xtan) {
  int wave = threadIdx.x >> 5;
  int lane = threadIdx.x & 31;
  int row  = blockIdx.x * 8 + wave;
  if (row >= ROWS_) return;
  const float* base = xseq + (size_t)row * DD_;
  float v[4]; float ss = 0.f;
#pragma unroll
  for (int u = 0; u < 4; ++u) { v[u] = base[1 + lane * 4 + u]; ss += v[u] * v[u]; }
#pragma unroll
  for (int off = 16; off > 0; off >>= 1) ss += __shfl_xor(ss, off, 32);
  float nrm = sqrtf(ss);
  float x0  = fmaxf(base[0], 1.0f + 1e-7f);
  float fac = acoshf(x0) / fmaxf(nrm, 1e-7f);
  float* out = xtan + (size_t)row * D_ + lane * 4;
#pragma unroll
  for (int u = 0; u < 4; ++u) out[u] = fac * v[u];
}

// =====================================================================
// K2: truncated DFT over T (15 modes).  One block per (b,j); threads = channel i.
// =====================================================================
__global__ void dft_kernel(const float* __restrict__ xtan,
                           float* __restrict__ xfr, float* __restrict__ xfi) {
  __shared__ float cs[T_ * M_];
  __shared__ float sn[T_ * M_];
  int bj = blockIdx.x;
  int i  = threadIdx.x;                      // 128 threads
  int b  = bj / J_, j = bj % J_;
  for (int idx = i; idx < T_ * M_; idx += 128) {
    int t = idx / M_, m = idx % M_;
    float th = 6.28318530717958647f * (float)(m * t) / (float)T_;
    __sincosf(th, &sn[idx], &cs[idx]);
  }
  __syncthreads();
  float ar[M_], ai[M_];
#pragma unroll
  for (int m = 0; m < M_; ++m) { ar[m] = 0.f; ai[m] = 0.f; }
  size_t base = ((size_t)b * T_ * J_ + j) * D_ + i;
  for (int t = 0; t < T_; ++t) {
    float x = xtan[base + (size_t)t * J_ * D_];
    const float* c = &cs[t * M_];
    const float* s = &sn[t * M_];
#pragma unroll
    for (int m = 0; m < M_; ++m) {
      ar[m] = fmaf(x,  c[m], ar[m]);
      ai[m] = fmaf(-x, s[m], ai[m]);
    }
  }
#pragma unroll
  for (int m = 0; m < M_; ++m) {
    xfr[((size_t)m * BJ_ + bj) * D_ + i] = ar[m];
    xfi[((size_t)m * BJ_ + bj) * D_ + i] = ai[m];
  }
}

// =====================================================================
// K3: per-mode complex GEMM om[m,bj,o] = sum_i xf[m,bj,i] * W[m,i,o]
// =====================================================================
__global__ void mode_gemm_kernel(const float* __restrict__ xfr, const float* __restrict__ xfi,
                                 const float* __restrict__ Wr,  const float* __restrict__ Wi,
                                 float* __restrict__ omr, float* __restrict__ omi) {
  int bid = blockIdx.x;
  int m = bid / BJ_, bj = bid % BJ_;
  int o = threadIdx.x;                      // 128 threads
  __shared__ float xr[D_], xi[D_];
  xr[o] = xfr[((size_t)m * BJ_ + bj) * D_ + o];
  xi[o] = xfi[((size_t)m * BJ_ + bj) * D_ + o];
  __syncthreads();
  float accr = 0.f, acci = 0.f;
  const float* wrm = Wr + (size_t)m * D_ * D_ + o;
  const float* wim = Wi + (size_t)m * D_ * D_ + o;
  for (int i = 0; i < D_; ++i) {
    float wr = wrm[(size_t)i * D_];
    float wi = wim[(size_t)i * D_];
    accr = fmaf(xr[i],  wr, accr);
    accr = fmaf(-xi[i], wi, accr);
    acci = fmaf(xr[i],  wi, acci);
    acci = fmaf(xi[i],  wr, acci);
  }
  omr[((size_t)m * BJ_ + bj) * D_ + o] = accr;
  omi[((size_t)m * BJ_ + bj) * D_ + o] = acci;
}

// =====================================================================
// K4: channel-mix GEMM z[row,o] = sum_i x_tan[row,i] * bw[o,i]
//     WMMA bf16; B (128x128 bf16, 32 KB) staged to LDS via TDM.
// =====================================================================
__global__ void bias_gemm_kernel(const float* __restrict__ xtan, const __bf16* __restrict__ bwT,
                                 float* __restrict__ hbuf) {
  __shared__ __bf16 Ab[16 * 128];
  __shared__ __bf16 Bst[128 * 128];
  int row0 = blockIdx.x * 16;
  int tid = threadIdx.x, wave = tid >> 5, lane = tid & 31;

  if (tid < 32) {   // wave 0 issues the DMA (TDM ignores EXEC; skipped waves never fetch it)
    tdm_load_2d_bf16(bwT, lds_offset_of(Bst), D_, D_, D_, D_, D_);
    __builtin_amdgcn_s_wait_tensorcnt(0);
  }
  for (int idx = tid; idx < 16 * 128; idx += 128)
    Ab[idx] = (__bf16)xtan[(size_t)(row0 + (idx >> 7)) * D_ + (idx & 127)];
  __syncthreads();

  int m = lane & 15, half = lane >> 4;
  for (int nt = 0; nt < 2; ++nt) {
    int n0 = (wave * 2 + nt) * 16;
    v8f acc = {};
#pragma unroll
    for (int kc = 0; kc < 4; ++kc) {
      int k0 = kc * 32;
      v16bf a  = a_frag(Ab, 128, m, half, k0);
      v16bf bm = b_frag(Bst, 128, k0 + lane, n0);
      acc = __builtin_amdgcn_wmma_f32_16x16x32_bf16(false, a, false, bm, (short)0, acc, false, false);
    }
    int n = n0 + (lane & 15);
#pragma unroll
    for (int v = 0; v < 8; ++v) {
      int mm = v + 8 * half;
      hbuf[(size_t)(row0 + mm) * D_ + n] = acc[v];
    }
  }
}

// =====================================================================
// K5: 15-mode inverse DFT + fno bias + p-adic hard ball + residual + LN1
// =====================================================================
__global__ void spec_ln1_kernel(const float* __restrict__ xtan, float* __restrict__ hbuf,
                                const float* __restrict__ omr, const float* __restrict__ omi,
                                const float* __restrict__ bbf,
                                const float* __restrict__ g1, const float* __restrict__ be1) {
  __shared__ float cs[M_], sn[M_];
  __shared__ float wsum[4], wsum2[4];
  int row = blockIdx.x;
  int o   = threadIdx.x;
  int wave = o >> 5, lane = o & 31;
  int b   = row / (T_ * J_);
  int rem = row % (T_ * J_);
  int t   = rem / J_;
  int j   = rem % J_;
  int bj  = b * J_ + j;
  if (o < M_) {
    float th = 6.28318530717958647f * (float)(o * t) / (float)T_;
    __sincosf(th, &sn[o], &cs[o]);
  }
  __syncthreads();
  float acc = omr[((size_t)0 * BJ_ + bj) * D_ + o];
#pragma unroll
  for (int m = 1; m < M_; ++m) {
    float re = omr[((size_t)m * BJ_ + bj) * D_ + o];
    float im = omi[((size_t)m * BJ_ + bj) * D_ + o];
    acc += 2.0f * (re * cs[m] - im * sn[m]);
  }
  float spec = acc * (1.0f / (float)T_);
  float fno  = hbuf[(size_t)row * D_ + o] + spec + bbf[o];
  float act  = (fabsf(fno) <= 1.0f) ? fno : 0.0f;   // padic_ball_hard, r = 3^0 = 1
  float pre  = xtan[(size_t)row * D_ + o] + act;
  float s = pre, s2 = pre * pre;
#pragma unroll
  for (int off = 16; off > 0; off >>= 1) { s += __shfl_xor(s, off, 32); s2 += __shfl_xor(s2, off, 32); }
  if (lane == 0) { wsum[wave] = s; wsum2[wave] = s2; }
  __syncthreads();
  float S  = wsum[0] + wsum[1] + wsum[2] + wsum[3];
  float S2 = wsum2[0] + wsum2[1] + wsum2[2] + wsum2[3];
  float mu  = S / (float)D_;
  float var = S2 / (float)D_ - mu * mu;
  float rstd = rsqrtf(var + 1e-5f);
  hbuf[(size_t)row * D_ + o] = (pre - mu) * rstd * g1[o] + be1[o];
}

// =====================================================================
// K6: fused FFN (fc1 -> gelu -> fc2) with TDM-staged weights + WMMA bf16,
//     + residual + LN2 + exp_map0 -> output (B,T,J,129)
//     LDS: Ab 4KB + A2 16KB + Bst 32KB + pre2 8KB ~= 60KB.
// =====================================================================
__global__ void ffn_ln2_expmap_kernel(const float* __restrict__ hbuf,
                                      const __bf16* __restrict__ w1, const float* __restrict__ b1,
                                      const __bf16* __restrict__ w2, const float* __restrict__ b2,
                                      const float* __restrict__ g2, const float* __restrict__ be2,
                                      float* __restrict__ out) {
  __shared__ __bf16 Ab[16 * 128];     // h tile (A for fc1)
  __shared__ __bf16 A2[16 * 512];     // gelu(fc1) tile (A for fc2) -- never hits HBM
  __shared__ __bf16 Bst[32 * 512];    // TDM weight stage (32KB), reused by fc1 & fc2
  __shared__ float  pre2[16 * 128];
  __shared__ float  muA[16], rsA[16], scA[16], chA[16];
  int row0 = blockIdx.x * 16;
  int tid = threadIdx.x, wave = tid >> 5, lane = tid & 31;
  int m = lane & 15, half = lane >> 4;
  u32 bst_off = lds_offset_of(Bst);

  for (int idx = tid; idx < 16 * 128; idx += 128)
    Ab[idx] = (__bf16)hbuf[(size_t)(row0 + (idx >> 7)) * D_ + (idx & 127)];
  __syncthreads();

  // ---- fc1: K=128 in 4 TDM-staged chunks (32 rows x 512 cols, 32KB each)
  v8f acc1[8];
#pragma unroll
  for (int nt = 0; nt < 8; ++nt) acc1[nt] = (v8f){};
  for (int kc = 0; kc < 4; ++kc) {
    if (tid < 32) {
      tdm_load_2d_bf16(w1 + (size_t)kc * 32 * HID_, bst_off, HID_, D_, HID_, HID_, 32);
      __builtin_amdgcn_s_wait_tensorcnt(0);
    }
    __syncthreads();                      // DMA visible to all waves
    v16bf a = a_frag(Ab, 128, m, half, kc * 32);
#pragma unroll
    for (int nt = 0; nt < 8; ++nt) {
      int n0 = wave * 128 + nt * 16;
      v16bf bm = b_frag(Bst, HID_, lane, n0);   // local k = lane within chunk
      acc1[nt] = __builtin_amdgcn_wmma_f32_16x16x32_bf16(false, a, false, bm, (short)0, acc1[nt], false, false);
    }
    __syncthreads();                      // all reads done before next DMA overwrites
  }
  // bias + exact gelu -> A2
  for (int nt = 0; nt < 8; ++nt) {
    int n = wave * 128 + nt * 16 + (lane & 15);
    float bias = b1[n];
#pragma unroll
    for (int v = 0; v < 8; ++v) {
      int mm = v + 8 * half;
      float x  = acc1[nt][v] + bias;
      float ge = 0.5f * x * (1.0f + erff(x * 0.70710678118654752f));
      A2[mm * HID_ + n] = (__bf16)ge;
    }
  }
  __syncthreads();

  // ---- fc2: K=512 in 4 TDM-staged chunks (128 rows x 128 cols, 32KB each)
  v8f acc2[2];
  acc2[0] = (v8f){}; acc2[1] = (v8f){};
  for (int kc = 0; kc < 4; ++kc) {
    if (tid < 32) {
      tdm_load_2d_bf16(w2 + (size_t)kc * 128 * D_, bst_off, D_, HID_, D_, D_, 128);
      __builtin_amdgcn_s_wait_tensorcnt(0);
    }
    __syncthreads();
#pragma unroll
    for (int ks = 0; ks < 4; ++ks) {
      int k0 = kc * 128 + ks * 32;
      v16bf a = a_frag(A2, HID_, m, half, k0);
#pragma unroll
      for (int nt = 0; nt < 2; ++nt) {
        int n0 = wave * 32 + nt * 16;
        v16bf bm = b_frag(Bst, D_, ks * 32 + lane, n0);
        acc2[nt] = __builtin_amdgcn_wmma_f32_16x16x32_bf16(false, a, false, bm, (short)0, acc2[nt], false, false);
      }
    }
    __syncthreads();
  }
  for (int nt = 0; nt < 2; ++nt) {
    int n = wave * 32 + nt * 16 + (lane & 15);
    float bias = b2[n];
#pragma unroll
    for (int v = 0; v < 8; ++v) {
      int mm = v + 8 * half;
      pre2[mm * D_ + n] = acc2[nt][v] + bias + hbuf[(size_t)(row0 + mm) * D_ + n];
    }
  }
  __syncthreads();

  // ---- LN2 (per-row stats by 16 threads)
  if (tid < 16) {
    float s = 0.f, s2 = 0.f;
    for (int c = 0; c < D_; ++c) { float x = pre2[tid * D_ + c]; s += x; s2 += x * x; }
    float mu  = s / (float)D_;
    float var = s2 / (float)D_ - mu * mu;
    muA[tid] = mu; rsA[tid] = rsqrtf(var + 1e-5f);
  }
  __syncthreads();
  for (int idx = tid; idx < 16 * 128; idx += 128) {
    int r = idx >> 7, c = idx & 127;
    pre2[idx] = (pre2[idx] - muA[r]) * rsA[r] * g2[c] + be2[c];
  }
  __syncthreads();

  // ---- exp_map0
  if (tid < 16) {
    float s = 0.f;
    for (int c = 0; c < D_; ++c) { float x = pre2[tid * D_ + c]; s += x * x; }
    float nr = sqrtf(s);
    scA[tid] = sinhf(nr) / fmaxf(nr, 1e-7f);
    chA[tid] = coshf(nr);
  }
  __syncthreads();
  for (int idx = tid; idx < 16 * 128; idx += 128) {
    int r = idx >> 7, c = idx & 127;
    out[(size_t)(row0 + r) * DD_ + 1 + c] = pre2[idx] * scA[r];
  }
  if (tid < 16) out[(size_t)(row0 + tid) * DD_] = chA[tid];
}

// =====================================================================
extern "C" void kernel_launch(void* const* d_in, const int* in_sizes, int n_in,
                              void* d_out, int out_size, void* d_ws, size_t ws_size,
                              hipStream_t stream) {
  const float*  xseq = (const float*)d_in[0];
  // d_in[1] = vel_seq : unused by the reference
  const double* wr   = (const double*)d_in[2];
  const double* wi   = (const double*)d_in[3];
  const double* bw   = (const double*)d_in[4];
  const double* bb   = (const double*)d_in[5];
  const float*  ln1g = (const float*)d_in[6];
  const float*  ln1b = (const float*)d_in[7];
  const float*  ln2g = (const float*)d_in[8];
  const float*  ln2b = (const float*)d_in[9];
  const float*  fc1w = (const float*)d_in[10];
  const float*  fc1b = (const float*)d_in[11];
  const float*  fc2w = (const float*)d_in[12];
  const float*  fc2b = (const float*)d_in[13];
  (void)in_sizes; (void)n_in; (void)out_size; (void)ws_size;

  char* ws = (char*)d_ws;
  size_t off = 0;
  auto alloc = [&](size_t bytes) -> char* {
    char* p = ws + off;
    off += (bytes + 255) & ~(size_t)255;
    return p;
  };
  float*  xtan = (float*)alloc((size_t)ROWS_ * D_ * sizeof(float));       // 67.7 MB
  float*  hbuf = (float*)alloc((size_t)ROWS_ * D_ * sizeof(float));       // 67.7 MB (z, then h)
  float*  xfr  = (float*)alloc((size_t)M_ * BJ_ * D_ * sizeof(float));
  float*  xfi  = (float*)alloc((size_t)M_ * BJ_ * D_ * sizeof(float));
  float*  omr  = (float*)alloc((size_t)M_ * BJ_ * D_ * sizeof(float));
  float*  omi  = (float*)alloc((size_t)M_ * BJ_ * D_ * sizeof(float));
  float*  Wr   = (float*)alloc((size_t)M_ * D_ * D_ * sizeof(float));
  float*  Wi   = (float*)alloc((size_t)M_ * D_ * D_ * sizeof(float));
  __bf16* bwT  = (__bf16*)alloc((size_t)D_ * D_ * sizeof(__bf16));
  float*  bbf  = (float*)alloc((size_t)D_ * sizeof(float));
  __bf16* w1   = (__bf16*)alloc((size_t)D_ * HID_ * sizeof(__bf16));
  __bf16* w2   = (__bf16*)alloc((size_t)HID_ * D_ * sizeof(__bf16));

  convert_weights_kernel<<<1024, 256, 0, stream>>>(wr, wi, bw, bb, fc1w, fc2w,
                                                   Wr, Wi, bwT, bbf, w1, w2);
  logmap_kernel<<<ROWS_ / 8, 256, 0, stream>>>(xseq, xtan);
  dft_kernel<<<BJ_, 128, 0, stream>>>(xtan, xfr, xfi);
  mode_gemm_kernel<<<M_ * BJ_, 128, 0, stream>>>(xfr, xfi, Wr, Wi, omr, omi);
  bias_gemm_kernel<<<ROWS_ / 16, 128, 0, stream>>>(xtan, bwT, hbuf);
  spec_ln1_kernel<<<ROWS_, 128, 0, stream>>>(xtan, hbuf, omr, omi, bbf, ln1g, ln1b);
  ffn_ln2_expmap_kernel<<<ROWS_ / 16, 128, 0, stream>>>(hbuf, w1, fc1b, w2, fc2b,
                                                        ln2g, ln2b, (float*)d_out);
}
